// GRU_72722386255954
// MI455X (gfx1250) — matrix operand order repaired and checked by reading, compiled, not run on hardware
//
#include <hip/hip_runtime.h>
#include <hip/hip_bf16.h>
#include <math.h>

// Problem constants (match reference)
#define T_SEQ 64
#define BATCH 64
#define VOCAB 32000
#define EDIM  512
#define HDIM  512
#define LAY   2
#define KKC   1024  // E + H

typedef _Float16 v16h __attribute__((ext_vector_type(16)));
typedef _Float16 v8h  __attribute__((ext_vector_type(8)));
typedef float    v8f  __attribute__((ext_vector_type(8)));

// ---- WMMA helpers ---------------------------------------------------------
// A-matrix (16x32 f16) lane layout (contiguous K, matching B):
//  lanes 0-15: M=lane, K=0..15
//  lanes16-31: M=lane-16, K=16..31
static __device__ inline v16h a_frag(const _Float16* rowk, int lane) {
  const int off = (lane >= 16) ? 16 : 0;
  v8h lo = *(const v8h*)(rowk + off);        // K = off + 0..7
  v8h hi = *(const v8h*)(rowk + 8 + off);    // K = off + 8..15
  return __builtin_shufflevector(lo, hi, 0,1,2,3,4,5,6,7,8,9,10,11,12,13,14,15);
}

// B-matrix (32x16 f16): lane holds column N=lane%16; lanes 0-15 hold K=0..15,
// lanes 16-31 hold K=16..31 (16 contiguous halves => one 32B vector load).
// Caller bakes the (lane>=16 ? +16 : 0) K offset into the base pointer.
static __device__ inline v8f mma(v16h a, v16h b, v8f c) {
  return __builtin_amdgcn_wmma_f32_16x16x32_f16(false, a, false, b, (short)0, c,
                                                false, false);
}

// Software-pipelined 16x16-tile GEMM segment: K = KSTEPS*32 halves.
// Loads for step s+1 are issued before the WMMA of step s so the scheduler
// can overlap global loads with v_wmma (partial s_wait_loadcnt).
template <int KSTEPS>
static __device__ inline v8f gemm_seg(const _Float16* __restrict__ arow,
                                      const _Float16* __restrict__ brow,
                                      int lane, v8f c) {
  v16h a0 = a_frag(arow, lane);
  v16h b0 = *(const v16h*)(brow);
#pragma unroll
  for (int s = 1; s < KSTEPS; ++s) {
    v16h a1 = a_frag(arow + s * 32, lane);
    v16h b1 = *(const v16h*)(brow + s * 32);
    c = mma(a0, b0, c);
    a0 = a1; b0 = b1;
  }
  return mma(a0, b0, c);
}

static __device__ inline float sigmoidf_(float x) {
  return 1.0f / (1.0f + __expf(-x));
}

// ---- Prep kernels ---------------------------------------------------------
__global__ __launch_bounds__(256) void k_conv_gates(
    const float* __restrict__ Wr, const float* __restrict__ Wz,
    const float* __restrict__ Wh, _Float16* __restrict__ dst) {
  const size_t i = (size_t)blockIdx.x * blockDim.x + threadIdx.x;
  const size_t total = (size_t)LAY * 3 * HDIM * KKC;
  if (i >= total) return;
  const size_t per_gate = (size_t)HDIM * KKC;
  const size_t layer  = i / (3 * per_gate);
  const size_t gate   = (i / per_gate) % 3;
  const size_t within = i % per_gate;
  const float* src = (gate == 0) ? Wr : (gate == 1) ? Wz : Wh;
  dst[i] = (_Float16)src[layer * per_gate + within];
}

__global__ __launch_bounds__(256) void k_conv_outw(
    const float* __restrict__ W, _Float16* __restrict__ dst) {
  const size_t i = (size_t)blockIdx.x * blockDim.x + threadIdx.x;
  if (i >= (size_t)VOCAB * HDIM) return;
  dst[i] = (_Float16)W[i];
}

__global__ __launch_bounds__(256) void k_embed(
    const int* __restrict__ tok, const float* __restrict__ emb,
    _Float16* __restrict__ dst) {
  const size_t i = (size_t)blockIdx.x * blockDim.x + threadIdx.x;
  if (i >= (size_t)T_SEQ * BATCH * EDIM) return;
  const size_t tb = i >> 9;          // / EDIM
  const size_t e  = i & (EDIM - 1);
  dst[i] = (_Float16)emb[(size_t)tok[tb] * EDIM + e];
}

__global__ __launch_bounds__(256) void k_inith(
    const float* __restrict__ h0, float* __restrict__ h32,
    _Float16* __restrict__ h16) {
  const size_t i = (size_t)blockIdx.x * blockDim.x + threadIdx.x;
  if (i >= (size_t)LAY * BATCH * HDIM) return;
  h32[i] = h0[i];
  h16[i] = (_Float16)h0[i];
}

__global__ __launch_bounds__(256) void k_final_hidden(
    const float* __restrict__ h32, float* __restrict__ out) {
  const size_t i = (size_t)blockIdx.x * blockDim.x + threadIdx.x;
  if (i >= (size_t)LAY * BATCH * HDIM) return;
  out[i] = h32[i];
}

// ---- GRU phase 1: r, z (K=1024), and x-part of candidate (K=512) ----------
// Latency-bound (serial chain) => spread wide: grid = (HDIM/16 = 32, 3 gates),
// block = 128 (4 waves; wave w owns rows w*16..w*16+15, one 16x16 tile).
__global__ __launch_bounds__(128) void k_gru_phase1(
    const _Float16* __restrict__ xsrc,  // [B, EDIM]
    const _Float16* __restrict__ hsrc,  // [B, HDIM] f16 hidden (prev)
    const _Float16* __restrict__ Wl,    // [3][H][KKC] f16, this layer
    const float* __restrict__ br, const float* __restrict__ bz,
    const float* __restrict__ bh,
    const float* __restrict__ h32,      // [B,H] f32 hidden (prev)
    _Float16* __restrict__ rh16, float* __restrict__ z32,
    float* __restrict__ hx32) {
  const int lane = threadIdx.x & 31;
  const int wave = threadIdx.x >> 5;   // 0..3 -> M tile
  const int gate = blockIdx.y;         // 0=r 1=z 2=hx
  const int row  = wave * 16 + (lane & 15);
  const int n    = blockIdx.x * 16 + (lane & 15);
  const int kbB  = (lane >= 16) ? 16 : 0;

  const _Float16* Wg   = Wl + (size_t)gate * HDIM * KKC;
  const _Float16* brow = Wg + (size_t)n * KKC + kbB;
  const _Float16* arow0 = xsrc + (size_t)row * EDIM;

  v8f c = {};
  c = gemm_seg<EDIM / 32>(arow0, brow, lane, c);                 // x segment
  if (gate < 2) {                                                // h segment
    const _Float16* arow1 = hsrc + (size_t)row * HDIM;
    c = gemm_seg<HDIM / 32>(arow1, brow + EDIM, lane, c);
  }

#pragma unroll
  for (int v = 0; v < 8; ++v) {
    const int m = wave * 16 + v + ((lane >= 16) ? 8 : 0);
    const size_t idx = (size_t)m * HDIM + n;
    const float acc = c[v];
    if (gate == 0) {
      const float r = sigmoidf_(acc + br[n]);
      rh16[idx] = (_Float16)(r * h32[idx]);
    } else if (gate == 1) {
      z32[idx] = sigmoidf_(acc + bz[n]);
    } else {
      hx32[idx] = acc + bh[n];
    }
  }
}

// ---- GRU phase 2: candidate (r*h part, K=512) + gate combine --------------
// grid = (HDIM/16 = 32), block = 128 (4 waves, one 16x16 tile each)
__global__ __launch_bounds__(128) void k_gru_phase2(
    const _Float16* __restrict__ rh16,  // [B,H]
    const _Float16* __restrict__ Wh2,   // layer's Wh rows, col offset E, row stride KKC
    const float* __restrict__ hx32, const float* __restrict__ z32,
    float* __restrict__ h32, _Float16* __restrict__ h16,
    _Float16* __restrict__ xout /* X[t] for top layer, else nullptr */) {
  const int lane = threadIdx.x & 31;
  const int wave = threadIdx.x >> 5;
  const int row  = wave * 16 + (lane & 15);
  const int n    = blockIdx.x * 16 + (lane & 15);
  const int kbB  = (lane >= 16) ? 16 : 0;

  const _Float16* brow = Wh2 + (size_t)n * KKC + kbB;
  const _Float16* arow = rh16 + (size_t)row * HDIM;

  v8f c = {};
  c = gemm_seg<HDIM / 32>(arow, brow, lane, c);

#pragma unroll
  for (int v = 0; v < 8; ++v) {
    const int m = wave * 16 + v + ((lane >= 16) ? 8 : 0);
    const size_t idx = (size_t)m * HDIM + n;
    const float hhat = tanhf(c[v] + hx32[idx]);
    const float z = z32[idx];
    const float hnew = (1.0f - z) * h32[idx] + z * hhat;
    h32[idx] = hnew;
    h16[idx] = (_Float16)hnew;
    if (xout) xout[idx] = (_Float16)hnew;
  }
}

// ---- Output projection: logits[4096,32000] = X @ out_w^T + b --------------
// grid = (VOCAB/64, 4096/128), block = 256 (8 waves; wave = 16x64 tile).
// Software-pipelined: fragments for k+32 are in flight during the 4 WMMAs of k.
__global__ __launch_bounds__(256) void k_proj(
    const _Float16* __restrict__ X,    // [4096, 512]
    const _Float16* __restrict__ W,    // [VOCAB, 512] f16
    const float* __restrict__ bias,    // [VOCAB]
    float* __restrict__ out) {         // [4096, VOCAB]
  const int lane = threadIdx.x & 31;
  const int wave = threadIdx.x >> 5;   // 0..7
  const int m0   = blockIdx.y * 128 + wave * 16;
  const int row  = m0 + (lane & 15);
  const int kbB  = (lane >= 16) ? 16 : 0;

  const _Float16* arow = X + (size_t)row * HDIM;
  const _Float16* bbase[4];
#pragma unroll
  for (int nt = 0; nt < 4; ++nt) {
    const int j = blockIdx.x * 64 + nt * 16 + (lane & 15);
    bbase[nt] = W + (size_t)j * HDIM + kbB;
    // Warm the L2->WGP path for the second half of each B row
    // (lowers to global_prefetch_b8, speculative).
    __builtin_prefetch(bbase[nt] + HDIM / 2, 0, 0);
  }

  v8f c[4] = {};
  // Pipeline prologue: fragments for k = 0
  v16h a0 = a_frag(arow, lane);
  v16h b0[4];
#pragma unroll
  for (int nt = 0; nt < 4; ++nt) b0[nt] = *(const v16h*)(bbase[nt]);

#pragma unroll
  for (int s = 1; s < HDIM / 32; ++s) {
    const int k = s * 32;
    v16h a1 = a_frag(arow + k, lane);
    v16h b1[4];
#pragma unroll
    for (int nt = 0; nt < 4; ++nt) b1[nt] = *(const v16h*)(bbase[nt] + k);
#pragma unroll
    for (int nt = 0; nt < 4; ++nt) c[nt] = mma(a0, b0[nt], c[nt]);
    a0 = a1;
#pragma unroll
    for (int nt = 0; nt < 4; ++nt) b0[nt] = b1[nt];
  }
#pragma unroll
  for (int nt = 0; nt < 4; ++nt) c[nt] = mma(a0, b0[nt], c[nt]);

#pragma unroll
  for (int nt = 0; nt < 4; ++nt) {
    const int n = blockIdx.x * 64 + nt * 16 + (lane & 15);
    const float bn = bias[n];
#pragma unroll
    for (int v = 0; v < 8; ++v) {
      const int m = m0 + v + ((lane >= 16) ? 8 : 0);
      out[(size_t)m * VOCAB + n] = c[nt][v] + bn;
    }
  }
}

// ---- Host side ------------------------------------------------------------
static inline size_t align256(size_t x) { return (x + 255) & ~(size_t)255; }

extern "C" void kernel_launch(void* const* d_in, const int* in_sizes, int n_in,
                              void* d_out, int out_size, void* d_ws,
                              size_t ws_size, hipStream_t stream) {
  (void)in_sizes; (void)n_in; (void)out_size; (void)ws_size;

  const int*   inputs = (const int*)d_in[0];
  const float* hidden = (const float*)d_in[1];
  const float* emb    = (const float*)d_in[2];
  const float* Wr_w   = (const float*)d_in[3];
  const float* Wr_b   = (const float*)d_in[4];
  const float* Wz_w   = (const float*)d_in[5];
  const float* Wz_b   = (const float*)d_in[6];
  const float* Wh_w   = (const float*)d_in[7];
  const float* Wh_b   = (const float*)d_in[8];
  const float* out_w  = (const float*)d_in[9];
  const float* out_b  = (const float*)d_in[10];
  float* out = (float*)d_out;

  // Workspace carve (all 256B-aligned): ~48.3 MB total
  char* p = (char*)d_ws;
  size_t off = 0;
  _Float16* W_h    = (_Float16*)(p + off); off = align256(off + (size_t)LAY*3*HDIM*KKC*2);
  _Float16* outw_h = (_Float16*)(p + off); off = align256(off + (size_t)VOCAB*HDIM*2);
  _Float16* xemb   = (_Float16*)(p + off); off = align256(off + (size_t)T_SEQ*BATCH*EDIM*2);
  _Float16* xout   = (_Float16*)(p + off); off = align256(off + (size_t)T_SEQ*BATCH*HDIM*2);
  float*    h32    = (float*)   (p + off); off = align256(off + (size_t)LAY*BATCH*HDIM*4);
  _Float16* h16    = (_Float16*)(p + off); off = align256(off + (size_t)LAY*BATCH*HDIM*2);
  _Float16* rh16   = (_Float16*)(p + off); off = align256(off + (size_t)BATCH*HDIM*2);
  float*    z32    = (float*)   (p + off); off = align256(off + (size_t)BATCH*HDIM*4);
  float*    hx32   = (float*)   (p + off); off = align256(off + (size_t)BATCH*HDIM*4);

  // Prologue: weight/embedding conversion + hidden init
  {
    const size_t ng = (size_t)LAY*3*HDIM*KKC;
    k_conv_gates<<<dim3((unsigned)((ng + 255)/256)), 256, 0, stream>>>(Wr_w, Wz_w, Wh_w, W_h);
    const size_t no = (size_t)VOCAB*HDIM;
    k_conv_outw<<<dim3((unsigned)((no + 255)/256)), 256, 0, stream>>>(out_w, outw_h);
    const size_t ne = (size_t)T_SEQ*BATCH*EDIM;
    k_embed<<<dim3((unsigned)((ne + 255)/256)), 256, 0, stream>>>(inputs, emb, xemb);
    const size_t nh = (size_t)LAY*BATCH*HDIM;
    k_inith<<<dim3((unsigned)((nh + 255)/256)), 256, 0, stream>>>(hidden, h32, h16);
  }

  // Sequential GRU recurrence: 2 small WMMA kernels per (t, layer)
  for (int t = 0; t < T_SEQ; ++t) {
    for (int layer = 0; layer < LAY; ++layer) {
      const _Float16* xsrc = (layer == 0) ? (xemb + (size_t)t*BATCH*EDIM)
                                          : h16;  // layer0's fresh output
      const _Float16* hsrc = h16 + (size_t)layer*BATCH*HDIM;
      const _Float16* Wl   = W_h + (size_t)layer*3*HDIM*KKC;
      const float* br = Wr_b + (size_t)layer*HDIM;
      const float* bz = Wz_b + (size_t)layer*HDIM;
      const float* bh = Wh_b + (size_t)layer*HDIM;
      float* h32l = h32 + (size_t)layer*BATCH*HDIM;

      k_gru_phase1<<<dim3(HDIM/16, 3), 128, 0, stream>>>(
          xsrc, hsrc, Wl, br, bz, bh, h32l, rh16, z32, hx32);

      const _Float16* Wh2 = Wl + (size_t)2*HDIM*KKC + EDIM;
      _Float16* xo = (layer == LAY-1) ? (xout + (size_t)t*BATCH*HDIM) : nullptr;
      k_gru_phase2<<<dim3(HDIM/16), 128, 0, stream>>>(
          rh16, Wh2, hx32, z32, h32l, h16 + (size_t)layer*BATCH*HDIM, xo);
    }
  }

  // Big projection GEMM: [T*B, V] = X @ out_w^T + b
  k_proj<<<dim3(VOCAB/64, (T_SEQ*BATCH)/128), 256, 0, stream>>>(
      xout, outw_h, out_b, out);

  // Final hidden -> tail of d_out
  {
    const size_t nh = (size_t)LAY*BATCH*HDIM;
    k_final_hidden<<<dim3((unsigned)((nh + 255)/256)), 256, 0, stream>>>(
        h32, out + (size_t)T_SEQ*BATCH*VOCAB);
  }
}
